// Inv1x1ConvPermute_72060961292861
// MI455X (gfx1250) — matrix operand order repaired and checked by reading, not compiled, not run on hardware
//
#include <hip/hip_runtime.h>
#include <cstdint>

typedef __bf16 bf16_t;
typedef __attribute__((ext_vector_type(16))) __bf16 v16bf;
typedef __attribute__((ext_vector_type(8)))  float  v8f;
typedef __attribute__((ext_vector_type(4)))  float  f32x4;

#define K_TOT 256
#define N_TOT 256
#define BM    128
#define BK    32
#define PITCH 36   // 32 data dwords + 4 pad: keeps 16B alignment, conflict-free ds_load_b128

__device__ __forceinline__ void split_f32(float f, __bf16 &hi, __bf16 &lo) {
    __bf16 h = (__bf16)f;          // RNE to bf16 (top ~8 mantissa bits)
    hi = h;
    lo = (__bf16)(f - (float)h);   // exact residual in fp32, next ~8 bits
}

// kernel [K][N] fp32 -> Bt_hi/Bt_lo [N][K] bf16 (transposed so each lane's
// WMMA B fragment is one contiguous 32B run).
__global__ void prep_b_kernel(const float* __restrict__ kern,
                              bf16_t* __restrict__ bhi,
                              bf16_t* __restrict__ blo) {
    const int n = blockIdx.x;
    const int k = threadIdx.x;
    __bf16 h, l;
    split_f32(kern[k * N_TOT + n], h, l);
    bhi[n * K_TOT + k] = h;
    blo[n * K_TOT + k] = l;
}

__global__ __launch_bounds__(256)
void gemm_bf16x3_kernel(const float* __restrict__ x,
                        const bf16_t* __restrict__ bt_hi,
                        const bf16_t* __restrict__ bt_lo,
                        float* __restrict__ out) {
    __shared__ __align__(16) float smem[2][BM * PITCH];

    const int tid  = threadIdx.x;
    const int lane = tid & 31;
    const int wave = tid >> 5;
    const int wm   = wave & 1;   // 64-row group of this wave
    const int wn   = wave >> 1;  // 64-col group of this wave
    const int l15  = lane & 15;
    const int lhi  = lane >> 4;

    const int64_t m0 = (int64_t)blockIdx.x * BM;

    // ---- cooperative async tile loader: each thread owns 16 floats (4 x b128)
    const int ldr = tid >> 1;           // row 0..127
    const int ldc = (tid & 1) * 16;     // col 0 or 16
    const float*   gsrc   = x + (m0 + ldr) * K_TOT + ldc;
    const uint32_t lds0   = (uint32_t)(uintptr_t)(&smem[0][0])
                          + (uint32_t)(ldr * PITCH + ldc) * 4u;
    const uint32_t ldsbuf = (uint32_t)(BM * PITCH) * 4u;

#define ISSUE_TILE(buf, k0)                                                   \
    do {                                                                      \
        const float* _g = gsrc + (k0);                                        \
        uint32_t _l = lds0 + (uint32_t)(buf) * ldsbuf;                        \
        _Pragma("unroll")                                                     \
        for (int _j = 0; _j < 4; ++_j) {                                      \
            uint64_t _ga = (uint64_t)(uintptr_t)(_g + 4 * _j);                \
            asm volatile("global_load_async_to_lds_b128 %0, %1, off"          \
                         :: "v"(_l + 16u * (uint32_t)_j), "v"(_ga)            \
                         : "memory");                                         \
        }                                                                     \
    } while (0)

    v8f acc[4][4];
    #pragma unroll
    for (int mt = 0; mt < 4; ++mt)
        #pragma unroll
        for (int nt = 0; nt < 4; ++nt)
            #pragma unroll
            for (int e = 0; e < 8; ++e)
                acc[mt][nt][e] = 0.0f;

    ISSUE_TILE(0, 0);
    asm volatile("s_wait_asynccnt 0" ::: "memory");
    __syncthreads();

    const int arow = wm * 64 + l15;   // + mt*16
    const int ac0  = lhi * 16;        // lanes 0-15: K 0..15; 16-31: 16..31

    #pragma unroll 2
    for (int ks = 0; ks < 8; ++ks) {
        const int cur = ks & 1;
        if (ks < 7) ISSUE_TILE(cur ^ 1, (ks + 1) * BK);

        // A fragments: fp32 from LDS -> bf16 hi/lo, contiguous k = 16*(lane/16) + s order, same as B
        v16bf ahi[4], alo[4];
        #pragma unroll
        for (int mt = 0; mt < 4; ++mt) {
            const float* sp = &smem[cur][(arow + mt * 16) * PITCH + ac0];
            f32x4 f0 = *(const f32x4*)(sp + 0);
            f32x4 f1 = *(const f32x4*)(sp + 4);
            f32x4 f2 = *(const f32x4*)(sp + 8);
            f32x4 f3 = *(const f32x4*)(sp + 12);
            #pragma unroll
            for (int e = 0; e < 4; ++e) {
                __bf16 h, l;
                split_f32(f0[e], h, l); ahi[mt][e]      = h; alo[mt][e]      = l;
                split_f32(f1[e], h, l); ahi[mt][4 + e]  = h; alo[mt][4 + e]  = l;
                split_f32(f2[e], h, l); ahi[mt][8 + e]  = h; alo[mt][8 + e]  = l;
                split_f32(f3[e], h, l); ahi[mt][12 + e] = h; alo[mt][12 + e] = l;
            }
        }

        // B fragments from L2-resident transposed weights; 3-term bf16 WMMA
        const int kb = ks * BK + lhi * 16;
        #pragma unroll
        for (int nt = 0; nt < 4; ++nt) {
            const size_t boff = (size_t)(wn * 64 + nt * 16 + l15) * K_TOT + kb;
            v16bf bh = *(const v16bf*)(bt_hi + boff);
            v16bf bl = *(const v16bf*)(bt_lo + boff);
            #pragma unroll
            for (int mt = 0; mt < 4; ++mt) {
                acc[mt][nt] = __builtin_amdgcn_wmma_f32_16x16x32_bf16(
                    false, ahi[mt], false, bh, (short)0, acc[mt][nt], false, false);
                acc[mt][nt] = __builtin_amdgcn_wmma_f32_16x16x32_bf16(
                    false, alo[mt], false, bh, (short)0, acc[mt][nt], false, false);
                acc[mt][nt] = __builtin_amdgcn_wmma_f32_16x16x32_bf16(
                    false, ahi[mt], false, bl, (short)0, acc[mt][nt], false, false);
            }
        }

        asm volatile("s_wait_asynccnt 0" ::: "memory");
        __syncthreads();
    }
#undef ISSUE_TILE

    // C/D layout: VGPR i -> M = i + 8*(lane>=16), N = lane&15
    const int col = wn * 64 + l15;
    #pragma unroll
    for (int mt = 0; mt < 4; ++mt) {
        const int64_t r0 = m0 + wm * 64 + mt * 16 + lhi * 8;
        #pragma unroll
        for (int nt = 0; nt < 4; ++nt) {
            float* op = out + r0 * N_TOT + col + nt * 16;
            #pragma unroll
            for (int i = 0; i < 8; ++i)
                op[(int64_t)i * N_TOT] = acc[mt][nt][i];
        }
    }
}

extern "C" void kernel_launch(void* const* d_in, const int* in_sizes, int n_in,
                              void* d_out, int out_size, void* d_ws, size_t ws_size,
                              hipStream_t stream) {
    const float* x    = (const float*)d_in[0];
    const float* kern = (const float*)d_in[1];
    float*       out  = (float*)d_out;

    bf16_t* bhi = (bf16_t*)d_ws;                       // 128 KB
    bf16_t* blo = bhi + (size_t)K_TOT * N_TOT;         // 128 KB

    prep_b_kernel<<<N_TOT, K_TOT, 0, stream>>>(kern, bhi, blo);

    const int M = in_sizes[0] / K_TOT;                 // 262144
    gemm_bf16x3_kernel<<<M / BM, 256, 0, stream>>>(x, bhi, blo, out);
}
